// GraphMatcher_40157944218249
// MI455X (gfx1250) — compile-verified, matrix operand order checked by reading, not run on hardware
//
#include <hip/hip_runtime.h>
#include <hip/hip_bf16.h>
#include <stdint.h>

// ---------------------------------------------------------------------------
// GraphMatcher on MI455X (gfx1250, wave32).
// - sf-projection GEMM (17.2 GF, 268 MB read) in bf16 WMMA, f32 accum:
//   HBM-bound (~11.5us @ 23.3 TB/s), compute far under the bf16 matrix roof.
// - sat tiles staged Global->LDS by the Tensor Data Mover (double-buffered
//   D# descriptors, TENSORcnt split-wait), TDM hardware padding gives
//   65-dword LDS rows (bank-conflict free) for free.
// - B-tiles converted f32->bf16 ONCE per slab, pre-swizzled into the WMMA
//   lane layout so fragment builds are pure ds_load_b128.
// - sim_map (256MB) never materialized; 4-corner bilinear dots hit an
//   L2-resident (134MB < 192MB) pixel-major normalized sf buffer.
// ---------------------------------------------------------------------------

typedef __attribute__((ext_vector_type(16))) __bf16 v16bf;
typedef __attribute__((ext_vector_type(8)))  float  v8f;
typedef __attribute__((ext_vector_type(4)))  unsigned int v4u;
typedef __attribute__((ext_vector_type(8)))  int v8i;
typedef __attribute__((ext_vector_type(4)))  int v4i;

#define Bn      4
#define Nn      256
#define INDIM   256
#define HID     128
#define Hs      256
#define Wss     256
#define HW      (Hs * Wss)
#define S_CAND  256      // 16x16 search grid
#define PIXT    64       // pixels per workgroup tile in sat GEMM
#define KT      32       // K-slab per WMMA stage
#define XPAD    (PIXT + 1)
#define NSLAB   (INDIM / KT)

// round-to-nearest-even f32 -> bf16 (as raw u16)
__device__ __forceinline__ unsigned short f2bf(float f) {
  union { float f; unsigned u; } c; c.f = f;
  unsigned u = c.u + 0x7FFFu + ((c.u >> 16) & 1u);
  return (unsigned short)(u >> 16);
}

// ---------------------------------------------------------------------------
// Kernel 0: one-shot W_sat f32 -> bf16 (hoists cvt out of the GEMM K-loop)
// ---------------------------------------------------------------------------
__global__ void k_wconv(const float* __restrict__ W, unsigned short* __restrict__ Wbf) {
  int i = blockIdx.x * 256 + threadIdx.x;
  if (i < HID * INDIM) Wbf[i] = f2bf(W[i]);
}

// ---------------------------------------------------------------------------
// Kernel 1: nf = normalize(node_features @ W_node + b_node)   [B,N,128] f32
// ---------------------------------------------------------------------------
__global__ void k_node_proj(const float* __restrict__ feat,
                            const float* __restrict__ Wn,
                            const float* __restrict__ bnod,
                            float* __restrict__ nf) {
  __shared__ float sfeat[INDIM];
  __shared__ float red[HID];
  int row = blockIdx.x;            // b*N + n
  int c   = threadIdx.x;           // 0..127 output channel
  const float* fp = feat + (size_t)row * INDIM;
  sfeat[c]       = fp[c];
  sfeat[c + 128] = fp[c + 128];
  __syncthreads();
  float acc = bnod[c];
#pragma unroll 8
  for (int k = 0; k < INDIM; ++k) acc += sfeat[k] * Wn[k * HID + c];
  red[c] = acc * acc;
  __syncthreads();
  for (int s = 64; s > 0; s >>= 1) {
    if (c < s) red[c] += red[c + s];
    __syncthreads();
  }
  float inv = 1.0f / fmaxf(sqrtf(red[0]), 1e-12f);
  nf[(size_t)row * HID + c] = acc * inv;
}

// ---------------------------------------------------------------------------
// Kernel 2: sf_norm[b, pixel, o] = normalize_o(W_sat @ sat[b,:,pixel] + b_sat)
// Per workgroup: 128(o) x 64(pixel) tile, K=256 in 8 slabs of 32.
// TDM double-buffer: wave 0 issues slab ks+1, s_wait_tensorcnt(1) covers
// slab ks. One cooperative f32->bf16 pass pre-swizzles B into WMMA lane
// order; 8 waves x 4 v_wmma_f32_16x16x32_bf16 per slab.
// ---------------------------------------------------------------------------
__global__ void k_sat_proj(const float* __restrict__ sat,
                           const unsigned short* __restrict__ Wbf,
                           const float* __restrict__ bsat,
                           float* __restrict__ sfn) {
  __shared__ unsigned short sW[HID * KT];             // A slab [m][k] bf16 (8 KB)
  __shared__ float sX[2][KT * XPAD];                  // TDM dest, padded rows (2x8.3KB)
  __shared__ __align__(32) unsigned short sB[4 * 32 * 16];  // B frags, lane layout (4KB)
  __shared__ float sOut[HID * XPAD];                  // result [o][p] f32 (33 KB)
  __shared__ float sInv[PIXT];

  int blk  = blockIdx.x;
  int b    = blk / (HW / PIXT);
  int p0   = (blk % (HW / PIXT)) * PIXT;
  int tid  = threadIdx.x;
  int lane = tid & 31;
  int wv   = tid >> 5;       // wave 0..7
  int m0   = wv * 16;        // this wave's M slice

  v8f acc0 = {}, acc1 = {}, acc2 = {}, acc3 = {};
  const float* satB = sat + (size_t)b * INDIM * HW + p0;

  // D# group1 (constant): data_size=4B, pad_enable, pad_interval=64 dwords,
  // pad_amount=1 dword; tensor_dim0=HW, tensor_dim1=INDIM; tile 64x32;
  // tensor_dim0_stride=HW.
  const v8i g1 = { (int)0x01520000, 0, (int)0x01000001, (int)0x00400000,
                   KT, (int)HW, 0, 0 };
  const v4i gz = { 0, 0, 0, 0 };

  auto tdm_issue = [&](float* dstLds, int k0) {
    unsigned lds_off = (unsigned)(size_t)(void*)dstLds;
    unsigned long long ga =
        (unsigned long long)(size_t)(const void*)(satB + (size_t)k0 * HW);
    v4u g0 = { 1u, lds_off, (unsigned)ga,
               ((unsigned)((ga >> 32) & 0x01FFFFFFull)) | 0x80000000u };
#if __clang_major__ >= 23
    __builtin_amdgcn_tensor_load_to_lds(g0, g1, gz, gz, (v8i){}, 0);
#else
    __builtin_amdgcn_tensor_load_to_lds(g0, g1, gz, gz, 0);
#endif
  };

  if (wv == 0) tdm_issue(sX[0], 0);

  for (int ks = 0; ks < NSLAB; ++ks) {
    int cur = ks & 1;
    int k0  = ks * KT;
    if (wv == 0) {
      if (ks + 1 < NSLAB) {
        tdm_issue(sX[cur ^ 1], k0 + KT);        // prefetch next slab
        __builtin_amdgcn_s_wait_tensorcnt(1);   // current slab complete
      } else {
        __builtin_amdgcn_s_wait_tensorcnt(0);
      }
    }
    __syncthreads();

    // --- stage W slab (bf16, pre-converted): 128x32 ushort via uint4 copies
    {
      int m = tid >> 1, half = tid & 1;
      const uint4* src = (const uint4*)(Wbf + m * INDIM + k0 + half * 16);
      uint4* dst = (uint4*)(sW + m * KT + half * 16);
      dst[0] = src[0];
      dst[1] = src[1];
    }
    // --- convert B slab ONCE, pre-swizzled into WMMA lane layout:
    //     sB[((t*32)+L)*16 + e] = bf16(sX[cur][k(L,e)*XPAD + t*16 + (L&15)])
    {
      const float* xc = sX[cur];
#pragma unroll
      for (int i = 0; i < 8; ++i) {
        int f = tid * 8 + i;
        int e = f & 15, L = (f >> 4) & 31, t = f >> 9;
        int k = ((e >> 3) << 4) + ((L >> 4) << 3) + (e & 7);
        int p = t * 16 + (L & 15);
        sB[f] = f2bf(xc[k * XPAD + p]);
      }
    }
    __syncthreads();

    // --- fragments: A via two contiguous 8-elem runs, B via ds_load_b128 x2
    int kb = (lane >> 4) * 8;
    int mr = m0 + (lane & 15);
    union { v16bf v; unsigned short s[16]; } A;
#pragma unroll
    for (int e = 0; e < 16; ++e) {
      int k = ((e >> 3) << 4) + kb + (e & 7);
      A.s[e] = sW[mr * KT + k];
    }
    v16bf B0 = *(const v16bf*)(sB + (0 * 32 + lane) * 16);
    v16bf B1 = *(const v16bf*)(sB + (1 * 32 + lane) * 16);
    v16bf B2 = *(const v16bf*)(sB + (2 * 32 + lane) * 16);
    v16bf B3 = *(const v16bf*)(sB + (3 * 32 + lane) * 16);

    acc0 = __builtin_amdgcn_wmma_f32_16x16x32_bf16(false, A.v, false, B0,
                                                   (short)0, acc0, false, false);
    acc1 = __builtin_amdgcn_wmma_f32_16x16x32_bf16(false, A.v, false, B1,
                                                   (short)0, acc1, false, false);
    acc2 = __builtin_amdgcn_wmma_f32_16x16x32_bf16(false, A.v, false, B2,
                                                   (short)0, acc2, false, false);
    acc3 = __builtin_amdgcn_wmma_f32_16x16x32_bf16(false, A.v, false, B3,
                                                   (short)0, acc3, false, false);
  }
  __syncthreads();

  // dump accumulators (+bias): lane L holds N=L&15, VGPR r -> M = r + 8*(L>>4)
  int nl = lane & 15;
  int mb = m0 + ((lane >> 4) << 3);
#pragma unroll
  for (int r = 0; r < 8; ++r) {
    int m = mb + r;
    float bv = bsat[m];
    sOut[m * XPAD + nl]      = acc0[r] + bv;
    sOut[m * XPAD + 16 + nl] = acc1[r] + bv;
    sOut[m * XPAD + 32 + nl] = acc2[r] + bv;
    sOut[m * XPAD + 48 + nl] = acc3[r] + bv;
  }
  __syncthreads();

  // per-pixel channel L2 norm (all 128 channels live in this workgroup)
  if (tid < PIXT) {
    float s = 0.f;
#pragma unroll 4
    for (int o = 0; o < HID; ++o) {
      float v = sOut[o * XPAD + tid];
      s += v * v;
    }
    sInv[tid] = 1.0f / fmaxf(sqrtf(s), 1e-12f);
  }
  __syncthreads();

  // pixel-major store: sfn[(b*HW + p)*128 + o]
  float* outB = sfn + ((size_t)b * HW + p0) * HID;
#pragma unroll
  for (int i = 0; i < 32; ++i) {
    int idx = tid * 32 + i;
    int p = idx >> 7, o = idx & 127;
    outB[(size_t)p * HID + o] = sOut[o * XPAD + p] * sInv[p];
  }
}

// ---------------------------------------------------------------------------
// Kernel 3: per (b,s) candidate: geometry + fused bilinear similarity.
// ---------------------------------------------------------------------------
__global__ void k_sample(const float* __restrict__ ncoords,
                         const float* __restrict__ nscores,
                         const float* __restrict__ ndepths,
                         const float* __restrict__ Kl,
                         const float* __restrict__ Rl,
                         const float* __restrict__ tinit,
                         const float* __restrict__ Ksat,
                         const float* __restrict__ Rsat,
                         const float* __restrict__ tsat,
                         const float* __restrict__ srad,
                         const float* __restrict__ nf,
                         const float* __restrict__ sfn,
                         float* __restrict__ cc,
                         float* __restrict__ logits,
                         float* __restrict__ out) {
  __shared__ float red[S_CAND];
  int blk = blockIdx.x;
  int b = blk >> 8;
  int s = blk & 255;
  int n = threadIdx.x;

  // candidate translation (linspace(-1,1,16) grid * radius)
  int sx = s & 15, sy = s >> 4;
  float r  = srad[b];
  float ox = (-1.f + 2.f * (float)sx / 15.f) * r;
  float oy = (-1.f + 2.f * (float)sy / 15.f) * r;
  float ctx = tinit[b * 3 + 0] + ox;
  float cty = tinit[b * 3 + 1] + oy;
  float ctz = tinit[b * 3 + 2];
  if (n == 0) {
    float* ct = out + 3072 + (size_t)(b * S_CAND + s) * 3;
    ct[0] = ctx; ct[1] = cty; ct[2] = ctz;
  }

  // K_left^-1 (adjugate)
  const float* K = Kl + b * 9;
  float a00=K[0],a01=K[1],a02=K[2],a10=K[3],a11=K[4],a12=K[5],a20=K[6],a21=K[7],a22=K[8];
  float det = a00*(a11*a22-a12*a21) - a01*(a10*a22-a12*a20) + a02*(a10*a21-a11*a20);
  float id  = 1.0f / det;
  float i00=(a11*a22-a12*a21)*id, i01=(a02*a21-a01*a22)*id, i02=(a01*a12-a02*a11)*id;
  float i10=(a12*a20-a10*a22)*id, i11=(a00*a22-a02*a20)*id, i12=(a02*a10-a00*a12)*id;
  float i20=(a10*a21-a11*a20)*id, i21=(a01*a20-a00*a21)*id, i22=(a00*a11-a01*a10)*id;

  // image2world
  float u = ncoords[((size_t)(b * Nn) + n) * 2 + 0];
  float v = ncoords[((size_t)(b * Nn) + n) * 2 + 1];
  float d = ndepths[b * Nn + n];
  float cx = (i00*u + i01*v + i02) * d;
  float cy = (i10*u + i11*v + i12) * d;
  float cz = (i20*u + i21*v + i22) * d;
  const float* R = Rl + b * 9;
  float wx = R[0]*cx + R[1]*cy + R[2]*cz + ctx;
  float wy = R[3]*cx + R[4]*cy + R[5]*cz + cty;
  float wz = R[6]*cx + R[7]*cy + R[8]*cz + ctz;

  // world2image (satellite): cam = R^T (world - t); pixel = K cam / z
  const float* Rt = Rsat + b * 9;
  float dx = wx - tsat[b*3+0], dy = wy - tsat[b*3+1], dz = wz - tsat[b*3+2];
  float sxc = Rt[0]*dx + Rt[3]*dy + Rt[6]*dz;
  float syc = Rt[1]*dx + Rt[4]*dy + Rt[7]*dz;
  float szc = Rt[2]*dx + Rt[5]*dy + Rt[8]*dz;
  const float* Km = Ksat + b * 9;
  float uu = Km[0]*sxc + Km[1]*syc + Km[2]*szc;
  float vv = Km[3]*sxc + Km[4]*syc + Km[5]*szc;
  float ww = Km[6]*sxc + Km[7]*syc + Km[8]*szc;
  float ccx = uu / ww, ccy = vv / ww;
  size_t ci = (((size_t)(b * S_CAND) + s) * Nn + n) * 2;
  cc[ci] = ccx; cc[ci + 1] = ccy;

  // grid_sample pixel coords (align_corners=False convention)
  float ncx = ccx / 255.f * 2.f - 1.f;
  float ncy = ccy / 255.f * 2.f - 1.f;
  float px = ((ncx + 1.f) * 256.f - 1.f) * 0.5f;
  float py = ((ncy + 1.f) * 256.f - 1.f) * 0.5f;
  float x0 = floorf(px), y0 = floorf(py);
  float wx1 = px - x0, wx0f = 1.f - wx1;
  float wy1 = py - y0, wy0f = 1.f - wy1;
  int xi0 = (int)x0, yi0 = (int)y0;

  const float4* nfp = (const float4*)(nf + ((size_t)(b * Nn) + n) * HID);
  float sim = 0.f;
  auto corner = [&](int xi, int yi, float w) {
    if (xi < 0 || xi > Wss - 1 || yi < 0 || yi > Hs - 1) return;
    const float4* sp =
        (const float4*)(sfn + ((size_t)b * HW + (size_t)yi * Wss + xi) * HID);
    float acc = 0.f;
#pragma unroll
    for (int q = 0; q < HID / 4; ++q) {
      float4 a = nfp[q], c = sp[q];
      acc += a.x * c.x + a.y * c.y + a.z * c.z + a.w * c.w;
    }
    sim += w * acc;
  };
  corner(xi0,     yi0,     wx0f * wy0f);
  corner(xi0 + 1, yi0,     wx1  * wy0f);
  corner(xi0,     yi0 + 1, wx0f * wy1);
  corner(xi0 + 1, yi0 + 1, wx1  * wy1);

  red[n] = nscores[b * Nn + n] * sim;
  __syncthreads();
  for (int t = 128; t > 0; t >>= 1) {
    if (n < t) red[n] += red[n + t];
    __syncthreads();
  }
  if (n == 0) logits[b * S_CAND + s] = red[0];
}

// ---------------------------------------------------------------------------
// Kernel 4: softmax over S, argmax (first-max), emit scores/match/valid.
// ---------------------------------------------------------------------------
__global__ void k_softmax_out(const float* __restrict__ logits,
                              const float* __restrict__ lscale,
                              const float* __restrict__ cc,
                              float* __restrict__ out) {
  __shared__ float red[S_CAND];
  __shared__ int   ridx[S_CAND];
  __shared__ float sM, sSum;
  __shared__ int   sBest;
  int b = blockIdx.x, t = threadIdx.x;
  float esc = __expf(lscale[0]);
  float l = logits[b * S_CAND + t] * esc;
  red[t] = l; ridx[t] = t;
  __syncthreads();
  for (int s2 = 128; s2 > 0; s2 >>= 1) {
    if (t < s2) {
      float o = red[t + s2];
      if (o > red[t] || (o == red[t] && ridx[t + s2] < ridx[t])) {
        red[t] = o; ridx[t] = ridx[t + s2];
      }
    }
    __syncthreads();
  }
  if (t == 0) { sM = red[0]; sBest = ridx[0]; }
  __syncthreads();
  float e = __expf(l - sM);
  red[t] = e;
  __syncthreads();
  for (int s2 = 128; s2 > 0; s2 >>= 1) {
    if (t < s2) red[t] += red[t + s2];
    __syncthreads();
  }
  if (t == 0) sSum = red[0];
  __syncthreads();
  out[6144 + b * S_CAND + t] = e / sSum;            // scores [B,S]

  int best = sBest;
  size_t ci = (((size_t)(b * S_CAND) + best) * Nn + t) * 2;
  float mx = cc[ci], my = cc[ci + 1];
  out[((size_t)(b * Nn) + t) * 2 + 0] = mx;         // match [B,N,2] at 0
  out[((size_t)(b * Nn) + t) * 2 + 1] = my;
  out[2048 + b * Nn + t] =                          // valid [B,N] at 2048
      (mx >= 0.f && mx < (float)Wss && my >= 0.f && my < (float)Hs) ? 1.0f : 0.0f;
}

// ---------------------------------------------------------------------------
extern "C" void kernel_launch(void* const* d_in, const int* in_sizes, int n_in,
                              void* d_out, int out_size, void* d_ws, size_t ws_size,
                              hipStream_t stream) {
  const float* node_coords   = (const float*)d_in[0];
  const float* node_scores   = (const float*)d_in[1];
  const float* node_features = (const float*)d_in[2];
  const float* node_depths   = (const float*)d_in[3];
  const float* K_left        = (const float*)d_in[4];
  const float* R_l2w         = (const float*)d_in[5];
  const float* t_init        = (const float*)d_in[6];
  const float* sat           = (const float*)d_in[7];
  const float* K_sat         = (const float*)d_in[8];
  const float* R_s2w         = (const float*)d_in[9];
  const float* t_sat         = (const float*)d_in[10];
  const float* srad          = (const float*)d_in[11];
  // d_in[12] = search_steps (16) baked into S_CAND
  const float* W_node        = (const float*)d_in[13];
  const float* b_node        = (const float*)d_in[14];
  const float* W_sat         = (const float*)d_in[15];
  const float* b_sat         = (const float*)d_in[16];
  const float* logit_scale   = (const float*)d_in[17];

  // workspace layout (f32): nf | sf_norm | cand_coords | logits | Wbf(bf16)
  float* ws  = (float*)d_ws;
  float* nf  = ws;                                        // B*N*128      = 131072
  float* sfn = ws + 131072;                               // B*HW*128     = 33554432
  float* cc  = sfn + (size_t)Bn * HW * HID;               // B*S*N*2      = 524288
  float* lg  = cc + (size_t)Bn * S_CAND * Nn * 2;         // B*S          = 1024
  unsigned short* wbf = (unsigned short*)(lg + 1024);     // 128*256 bf16
  float* out = (float*)d_out;

  k_wconv<<<(HID * INDIM + 255) / 256, 256, 0, stream>>>(W_sat, wbf);
  k_node_proj<<<Bn * Nn, 128, 0, stream>>>(node_features, W_node, b_node, nf);
  k_sat_proj<<<Bn * (HW / PIXT), 256, 0, stream>>>(sat, wbf, b_sat, sfn);
  k_sample<<<Bn * S_CAND, 256, 0, stream>>>(node_coords, node_scores, node_depths,
                                            K_left, R_l2w, t_init, K_sat, R_s2w,
                                            t_sat, srad, nf, sfn, cc, lg, out);
  k_softmax_out<<<Bn, 256, 0, stream>>>(lg, logit_scale, cc, out);
}
